// NRQM_41377714930114
// MI455X (gfx1250) — compile-verified, matrix operand order checked by reading, not run on hardware
//
#include <hip/hip_runtime.h>
#include <hip/hip_fp16.h>

typedef __attribute__((ext_vector_type(2)))  float        v2f;
typedef __attribute__((ext_vector_type(8)))  float        v8f;
typedef __attribute__((ext_vector_type(16))) _Float16     v16h;
typedef __attribute__((ext_vector_type(4)))  unsigned int u32x4;
typedef __attribute__((ext_vector_type(4)))  int          i32x4;
typedef __attribute__((ext_vector_type(8)))  int          i32x8;

#define BATCH 32
#define H1 960
#define W1 960
#define H2 480
#define W2 480
#define NB 100
#define NF 36

#if __has_builtin(__builtin_amdgcn_tensor_load_to_lds)
#define USE_TDM 1
#else
#define USE_TDM 0
#endif

// ---------------------------------------------------------------------------
// MSCN normalization: mu = gauss7(img), m2 = gauss7(img^2), replicate pad.
// One fused pass over an LDS tile (16x16 output + 3-halo = 22x22).
// ---------------------------------------------------------------------------
__global__ __launch_bounds__(256) void mscn_kernel(const float* __restrict__ in,
                                                   float* __restrict__ out,
                                                   int H, int W) {
  __shared__ float tile[22 * 22];
  int b = blockIdx.z;
  int ox = blockIdx.x * 16, oy = blockIdx.y * 16;
  const float* img = in + (size_t)b * H * W;
  int tid = threadIdx.y * 16 + threadIdx.x;
  for (int e = tid; e < 22 * 22; e += 256) {
    int r = e / 22, c = e % 22;
    int gy = oy + r - 3; gy = gy < 0 ? 0 : (gy >= H ? H - 1 : gy);
    int gx = ox + c - 3; gx = gx < 0 ? 0 : (gx >= W ? W - 1 : gx);
    tile[e] = img[(size_t)gy * W + gx];
  }
  __syncthreads();
  // separable 7-tap gaussian weights, sigma = 7/6  (1/(2*sigma^2) = 18/49)
  float wr[7];
  float s = 0.f;
  #pragma unroll
  for (int d = -3; d <= 3; ++d) { float w = __expf(-(float)(d * d) * (18.f / 49.f)); wr[d + 3] = w; s += w; }
  float invs = 1.f / s;
  #pragma unroll
  for (int i = 0; i < 7; ++i) wr[i] *= invs;

  int ty = threadIdx.y, tx = threadIdx.x;
  float mu = 0.f, m2 = 0.f;
  #pragma unroll
  for (int i = 0; i < 7; ++i) {
    float rv = 0.f, r2 = 0.f;
    #pragma unroll
    for (int j = 0; j < 7; ++j) {
      float v = tile[(ty + i) * 22 + tx + j];
      rv += wr[j] * v;
      r2 += wr[j] * v * v;
    }
    mu += wr[i] * rv;
    m2 += wr[i] * r2;
  }
  float v0  = tile[(ty + 3) * 22 + tx + 3];
  float sig = sqrtf(fabsf(m2 - mu * mu));
  out[(size_t)b * H * W + (size_t)(oy + ty) * W + (ox + tx)] = (v0 - mu) / (sig + 1.f);
}

// ---------------------------------------------------------------------------
// 3x3 gaussian (sigma 0.5, zero pad) + stride-2 decimation: 960 -> 480.
// ---------------------------------------------------------------------------
__global__ __launch_bounds__(256) void down_kernel(const float* __restrict__ in,
                                                   float* __restrict__ out) {
  int b = blockIdx.z;
  int x = blockIdx.x * 16 + threadIdx.x;
  int y = blockIdx.y * 16 + threadIdx.y;
  const float* img = in + (size_t)b * H1 * W1;
  float e2 = __expf(-2.f);               // exp(-d^2/(2*0.25)) for d=1
  float S  = 1.f + 2.f * e2;
  float w[3] = { e2 / S, 1.f / S, e2 / S };
  float acc = 0.f;
  #pragma unroll
  for (int dy = -1; dy <= 1; ++dy) {
    int yy = 2 * y + dy;
    if (yy < 0 || yy >= H1) continue;
    float rowa = 0.f;
    #pragma unroll
    for (int dx = -1; dx <= 1; ++dx) {
      int xx = 2 * x + dx;
      if (xx < 0 || xx >= W1) continue;
      rowa += w[dx + 1] * img[(size_t)yy * W1 + xx];
    }
    acc += w[dy + 1] * rowa;
  }
  out[(size_t)b * H2 * W2 + (size_t)y * W2 + x] = acc;
}

// ---------------------------------------------------------------------------
// AGGD feature extraction. One workgroup per (image, block, scale).
// Block tile (bs x bs strided out of W x W) is staged into LDS with the
// Tensor Data Mover when available; reductions run out of LDS.
// RGAM lookup regenerated on the fly (monotone) via lgammaf + bisection
// over the exact reference grid alpha = 0.2 + 0.001*i, i in [0, 9800].
// ---------------------------------------------------------------------------
__device__ __forceinline__ float aggd_rfun(float g) {
  return __expf(2.f * lgammaf(2.f / g) - lgammaf(1.f / g) - lgammaf(3.f / g));
}

__global__ __launch_bounds__(256) void aggd_kernel(const float* __restrict__ xn,
                                                   float* __restrict__ dfeat,
                                                   int bs, int W, int featOff) {
  __shared__ float smem[9216];  // tile (<= 96*96), then reused as 30x256 reduce scratch
  int n = blockIdx.x, b = blockIdx.y;
  int by = n / 10, bx = n % 10;
  const float* base = xn + (size_t)b * W * W + (size_t)by * bs * W + (size_t)bx * bs;
  int tid = threadIdx.x;
  int npix = bs * bs;

#if USE_TDM
  // ---- TDM: 2D strided tile (bs rows of bs floats, row stride W) -> packed LDS
  if (tid == 0) {
    unsigned long long gaddr = (unsigned long long)(const void*)base;
    unsigned int lds_lo = (unsigned int)(unsigned long long)(void*)&smem[0];
    u32x4 g0;
    g0[0] = 1u;                                            // count=1, user descriptor
    g0[1] = lds_lo;                                        // lds_addr
    g0[2] = (unsigned int)(gaddr & 0xffffffffu);           // global_addr[31:0]
    g0[3] = (unsigned int)((gaddr >> 32) & 0x1ffffffu)     // global_addr[56:32]
            | (2u << 30);                                  // type = 2 ("image")
    i32x8 g1;
    g1[0] = (2 << 16);                                     // data_size = 4B; wg_mask=0
    g1[1] = (int)((W & 0xffff) << 16);                     // tensor_dim0[15:0]
    g1[2] = (int)(((unsigned)W >> 16) | ((unsigned)(W & 0xffff) << 16)); // dim0 hi | dim1 lo
    g1[3] = (int)(((unsigned)W >> 16) | ((unsigned)(bs & 0xffff) << 16)); // dim1 hi | tile_dim0
    g1[4] = (int)(bs & 0xffff);                            // tile_dim1; tile_dim2 = 0
    g1[5] = (int)W;                                        // tensor_dim0_stride[31:0]
    g1[6] = 0;                                             // stride hi | dim1_stride lo
    g1[7] = 0;
    i32x4 z4 = {0, 0, 0, 0};
#if defined(__clang_major__) && (__clang_major__ >= 23)
    i32x8 z8 = {0, 0, 0, 0, 0, 0, 0, 0};
    __builtin_amdgcn_tensor_load_to_lds(g0, g1, z4, z4, z8, 0);
#else
    __builtin_amdgcn_tensor_load_to_lds(g0, g1, z4, z4, 0);
#endif
  }
  __builtin_amdgcn_s_wait_tensorcnt(0);
  __syncthreads();
#else
  for (int e = tid; e < npix; e += 256) {
    int i = e / bs, j = e % bs;
    smem[e] = base[(size_t)i * W + j];
  }
  __syncthreads();
#endif

  float acc[30];
  #pragma unroll
  for (int a = 0; a < 30; ++a) acc[a] = 0.f;

  for (int e = tid; e < npix; e += 256) {
    int i = e / bs, j = e % bs;
    float x0 = smem[e];
    #pragma unroll
    for (int s = 0; s < 5; ++s) {
      const int dis[5] = {0, 0, 1, 1, 1};
      const int djs[5] = {0, 1, 0, 1, -1};
      float v;
      if (s == 0) {
        v = x0;
      } else {
        int ii = i - dis[s]; if (ii < 0) ii += bs;
        int jj = j - djs[s]; if (jj < 0) jj += bs; if (jj >= bs) jj -= bs;
        v = x0 * smem[ii * bs + jj];
      }
      float v2 = v * v;
      acc[s * 6 + 0] += (v < 0.f) ? 1.f : 0.f;
      acc[s * 6 + 1] += (v > 0.f) ? 1.f : 0.f;
      acc[s * 6 + 2] += (v < 0.f) ? v2 : 0.f;
      acc[s * 6 + 3] += (v > 0.f) ? v2 : 0.f;
      acc[s * 6 + 4] += fabsf(v);
      acc[s * 6 + 5] += v2;
    }
  }
  __syncthreads();  // tile data no longer needed, reuse LDS
  #pragma unroll
  for (int a = 0; a < 30; ++a) smem[a * 256 + tid] = acc[a];
  __syncthreads();
  for (int str = 128; str > 0; str >>= 1) {
    if (tid < str) {
      #pragma unroll
      for (int a = 0; a < 30; ++a) smem[a * 256 + tid] += smem[a * 256 + tid + str];
    }
    __syncthreads();
  }

  if (tid < 5) {
    int s = tid;
    float cntL = smem[(s * 6 + 0) * 256], cntR = smem[(s * 6 + 1) * 256];
    float sL2  = smem[(s * 6 + 2) * 256], sR2  = smem[(s * 6 + 3) * 256];
    float sAbs = smem[(s * 6 + 4) * 256], sSq  = smem[(s * 6 + 5) * 256];
    float cl = fmaxf(cntL, 1.f), cr = fmaxf(cntR, 1.f);
    float lstd = sqrtf(sL2 / cl), rstd = sqrtf(sR2 / cr);
    float gh   = lstd / fmaxf(rstd, 1e-12f);
    float invn = 1.f / (float)npix;
    float mAbs = sAbs * invn, mSq = sSq * invn;
    float rhat = mAbs * mAbs / fmaxf(mSq, 1e-12f);
    float gh2  = gh * gh;
    float rhn  = rhat * (gh2 * gh + 1.f) * (gh + 1.f) / ((gh2 + 1.f) * (gh2 + 1.f));
    // monotone bisection over the reference grid
    int lo = 0, hi = 9800;
    while (lo < hi) {
      int mid = (lo + hi) >> 1;
      float g = 0.2f + 0.001f * (float)mid;
      if (aggd_rfun(g) < rhn) lo = mid + 1; else hi = mid;
    }
    int idx = lo;
    if (lo > 0) {
      float rHere = aggd_rfun(0.2f + 0.001f * (float)lo);
      float rPrev = aggd_rfun(0.2f + 0.001f * (float)(lo - 1));
      if (fabsf(rPrev - rhn) < fabsf(rHere - rhn)) idx = lo - 1;
    }
    float alpha = 0.2f + 0.001f * (float)idx;
    float conv  = __expf(0.5f * (lgammaf(1.f / alpha) - lgammaf(3.f / alpha)));
    float bl = lstd * conv, br = rstd * conv;
    float* dst = dfeat + ((size_t)b * NB + n) * NF + featOff;
    if (s == 0) {
      dst[0] = alpha;
      dst[1] = 0.5f * (bl + br);
    } else {
      float mean = (br - bl) * __expf(lgammaf(2.f / alpha) - lgammaf(1.f / alpha));
      int o = 2 + 4 * (s - 1);
      dst[o] = alpha; dst[o + 1] = mean; dst[o + 2] = bl; dst[o + 3] = br;
    }
  }
}

// ---------------------------------------------------------------------------
// Per-image feature mean + centering.
// ---------------------------------------------------------------------------
__global__ __launch_bounds__(256) void center_kernel(const float* __restrict__ dfeat,
                                                     float* __restrict__ cbuf,
                                                     float* __restrict__ mu_dist) {
  __shared__ float mu[NF];
  int b = blockIdx.x, tid = threadIdx.x;
  if (tid < NF) {
    float s = 0.f;
    for (int n = 0; n < NB; ++n) s += dfeat[((size_t)b * NB + n) * NF + tid];
    float m = s * (1.f / (float)NB);
    mu[tid] = m;
    mu_dist[b * NF + tid] = m;
  }
  __syncthreads();
  for (int e = tid; e < NB * NF; e += 256) {
    int f = e % NF;
    cbuf[(size_t)b * NB * NF + e] = dfeat[(size_t)b * NB * NF + e] - mu[f];
  }
}

// ---------------------------------------------------------------------------
// Covariance via WMMA: cov = c^T c, c is [100 x 36] per image, padded to 48.
// One wave per (image, tileM, tileN); fp32 16x16x4 WMMA, K fully unrolled.
// Then blend with cov_pris: covhalf = (cov_pris + cov/(NB-1)) / 2.
// ---------------------------------------------------------------------------
__device__ __forceinline__ float cval(const float* __restrict__ c, int b, int k, int f) {
  bool ok = (f < NF) && (k < NB);
  int ff = ok ? f : 0;
  int kk = ok ? k : 0;
  float v = c[((size_t)b * NB + kk) * NF + ff];
  return ok ? v : 0.f;
}

__global__ __launch_bounds__(32) void cov_wmma_kernel(const float* __restrict__ cbuf,
                                                      const float* __restrict__ cov_pris,
                                                      float* __restrict__ covhalf) {
  int tm = blockIdx.x, tn = blockIdx.y, b = blockIdx.z;
  int lane = threadIdx.x;
  int half = lane >> 4, l16 = lane & 15;
  int row = tm * 16 + l16;  // M index (A = c^T tile)
  int col = tn * 16 + l16;  // N index (B = c tile)
  v8f accv = {};
#if __has_builtin(__builtin_amdgcn_wmma_f32_16x16x4_f32)
  // fp32 WMMA path: A 16x4 layout -> lanes 0-15 hold K = k0,k0+1; lanes 16-31 K = k0+2,k0+3
  #pragma unroll
  for (int k0 = 0; k0 < NB; k0 += 4) {
    int ka = k0 + 2 * half;
    v2f A, Bv;
    A.x  = cval(cbuf, b, ka,     row);
    A.y  = cval(cbuf, b, ka + 1, row);
    Bv.x = cval(cbuf, b, ka,     col);
    Bv.y = cval(cbuf, b, ka + 1, col);
    accv = __builtin_amdgcn_wmma_f32_16x16x4_f32(false, A, false, Bv, (short)0, accv,
                                                 false, false);
  }
#else
  // fallback: f16 inputs, f32 accumulate, K padded 100 -> 128
  #pragma unroll
  for (int k0 = 0; k0 < 128; k0 += 32) {
    v16h A, Bv;
    #pragma unroll
    for (int t = 0; t < 8; ++t) {
      int kb = (t < 4) ? (2 * t + 8 * half) : (16 + 2 * (t - 4) + 8 * half);
      A[2 * t]      = (_Float16)cval(cbuf, b, k0 + kb,     row);
      A[2 * t + 1]  = (_Float16)cval(cbuf, b, k0 + kb + 1, row);
      Bv[2 * t]     = (_Float16)cval(cbuf, b, k0 + kb,     col);
      Bv[2 * t + 1] = (_Float16)cval(cbuf, b, k0 + kb + 1, col);
    }
    accv = __builtin_amdgcn_wmma_f32_16x16x32_f16(false, A, false, Bv, (short)0, accv,
                                                  false, false);
  }
#endif
  // D layout: VGPR v, lanes 0-15 -> (M=v, N=lane); lanes 16-31 -> (M=v+8, N=lane-16)
  #pragma unroll
  for (int v = 0; v < 8; ++v) {
    int r  = tm * 16 + v + (half ? 8 : 0);
    int cc = tn * 16 + l16;
    if (r < NF && cc < NF) {
      covhalf[((size_t)b * NF + r) * NF + cc] =
          0.5f * (cov_pris[r * NF + cc] + accv[v] * (1.f / (float)(NB - 1)));
    }
  }
}

// ---------------------------------------------------------------------------
// 36x36 SPD inverse (Gauss-Jordan, pinv of SPD == inverse) + Mahalanobis.
// One workgroup per image.
// ---------------------------------------------------------------------------
__global__ __launch_bounds__(128) void solve_kernel(const float* __restrict__ covhalf,
                                                    const float* __restrict__ mu_pris,
                                                    const float* __restrict__ mu_dist,
                                                    float* __restrict__ out) {
  __shared__ float Aug[36 * 72];
  __shared__ float fcol[36];
  __shared__ float diff[36];
  __shared__ float red[128];
  __shared__ float pivs;
  int b = blockIdx.x, tid = threadIdx.x;
  for (int e = tid; e < 36 * 72; e += 128) {
    int r = e / 72, cc = e % 72;
    Aug[e] = (cc < 36) ? covhalf[((size_t)b * 36 + r) * 36 + cc]
                       : ((cc - 36 == r) ? 1.f : 0.f);
  }
  __syncthreads();
  for (int p = 0; p < 36; ++p) {
    if (tid == 0) {
      float d = Aug[p * 72 + p];
      pivs = 1.f / ((d == 0.f) ? 1e-20f : d);
    }
    __syncthreads();
    float pv = pivs;
    for (int cc = tid; cc < 72; cc += 128) Aug[p * 72 + cc] *= pv;
    __syncthreads();
    for (int r = tid; r < 36; r += 128) fcol[r] = Aug[r * 72 + p];
    __syncthreads();
    for (int e = tid; e < 36 * 72; e += 128) {
      int r = e / 72, cc = e % 72;
      if (r != p) Aug[e] -= fcol[r] * Aug[p * 72 + cc];
    }
    __syncthreads();
  }
  if (tid < 36) diff[tid] = mu_pris[tid] - mu_dist[b * 36 + tid];
  __syncthreads();
  float q = 0.f;
  for (int e = tid; e < 36 * 36; e += 128) {
    int r = e / 36, cc = e % 36;
    q += diff[r] * Aug[r * 72 + 36 + cc] * diff[cc];
  }
  red[tid] = q;
  __syncthreads();
  for (int s = 64; s > 0; s >>= 1) {
    if (tid < s) red[tid] += red[tid + s];
    __syncthreads();
  }
  if (tid == 0) out[b] = sqrtf(fmaxf(red[0], 0.f));
}

// ---------------------------------------------------------------------------
extern "C" void kernel_launch(void* const* d_in, const int* in_sizes, int n_in,
                              void* d_out, int out_size, void* d_ws, size_t ws_size,
                              hipStream_t stream) {
  (void)in_sizes; (void)n_in; (void)out_size; (void)ws_size;
  const float* X        = (const float*)d_in[0];  // [32,1,960,960]
  const float* mu_pris  = (const float*)d_in[1];  // [36]
  const float* cov_pris = (const float*)d_in[2];  // [36,36]
  float* out = (float*)d_out;                     // [32]

  char* ws = (char*)d_ws;
  const size_t N1 = (size_t)BATCH * H1 * W1 * sizeof(float);  // 117,964,800
  const size_t N2 = (size_t)BATCH * H2 * W2 * sizeof(float);  //  29,491,200
  const size_t ND = (size_t)BATCH * NB * NF * sizeof(float);  //     460,800
  float* xn1     = (float*)(ws);
  float* img2    = (float*)(ws + N1);
  float* xn2     = (float*)(ws + N1 + N2);
  float* dfeat   = (float*)(ws + N1 + 2 * N2);
  float* cbuf    = (float*)(ws + N1 + 2 * N2 + ND);
  float* mu_dist = (float*)(ws + N1 + 2 * N2 + 2 * ND);
  float* covhalf = (float*)(ws + N1 + 2 * N2 + 2 * ND + 4608);

  dim3 blk(16, 16);
  mscn_kernel<<<dim3(W1 / 16, H1 / 16, BATCH), blk, 0, stream>>>(X, xn1, H1, W1);
  down_kernel<<<dim3(W2 / 16, H2 / 16, BATCH), blk, 0, stream>>>(X, img2);
  mscn_kernel<<<dim3(W2 / 16, H2 / 16, BATCH), blk, 0, stream>>>(img2, xn2, H2, W2);
  aggd_kernel<<<dim3(NB, BATCH), 256, 0, stream>>>(xn1, dfeat, 96, W1, 0);
  aggd_kernel<<<dim3(NB, BATCH), 256, 0, stream>>>(xn2, dfeat, 48, W2, 18);
  center_kernel<<<BATCH, 256, 0, stream>>>(dfeat, cbuf, mu_dist);
  cov_wmma_kernel<<<dim3(3, 3, BATCH), 32, 0, stream>>>(cbuf, cov_pris, covhalf);
  solve_kernel<<<BATCH, 128, 0, stream>>>(covhalf, mu_pris, mu_dist, out);
}
